// MultiHeadAttention_3032246911547
// MI455X (gfx1250) — matrix operand order repaired and checked by reading, not compiled, not run on hardware
//
#include <hip/hip_runtime.h>
#include <stdint.h>

// ---------------------------------------------------------------------------
// MHA forward for MI455X (gfx1250): bf16 WMMA GEMMs + flash-attention core.
// B=4, S=2048, D=1024, H=16, HD=64.
// ---------------------------------------------------------------------------

#define BB  4
#define SS  2048
#define DD  1024
#define HH  16
#define HDD 64

typedef __attribute__((ext_vector_type(16))) __bf16 v16bf;
typedef __attribute__((ext_vector_type(8)))  float  v8f;

union ARegs { uint4 u[2]; v16bf v; };   // 16 bf16 = 32 bytes = 8 VGPRs

__device__ __forceinline__ v8f wmma_bf16(const ARegs& a, const ARegs& b, v8f c) {
  // v_wmma_f32_16x16x32_bf16: D = A(16x32) * B(32x16) + C(16x16 f32)
  return __builtin_amdgcn_wmma_f32_16x16x32_bf16(
      /*neg_a=*/false, a.v, /*neg_b=*/false, b.v,
      /*c_mod=*/(short)0, c, /*reuse_a=*/false, /*reuse_b=*/false);
}

__device__ __forceinline__ unsigned short f2bf(float f) {
  unsigned int u = __builtin_bit_cast(unsigned int, f);
  u += 0x7FFFu + ((u >> 16) & 1u);       // round-to-nearest-even
  return (unsigned short)(u >> 16);
}

__device__ __forceinline__ v8f vzero8() {
  v8f z;
#pragma unroll
  for (int i = 0; i < 8; ++i) z[i] = 0.0f;
  return z;
}

// ---------------------------------------------------------------------------
// Stage 1: fp32 -> bf16 conversion (vectorized: b128 load, b64 store)
// ---------------------------------------------------------------------------
__global__ __launch_bounds__(256)
void cvt_f32_bf16(const float* __restrict__ src, unsigned short* __restrict__ dst, int n4) {
  int i = blockIdx.x * blockDim.x + threadIdx.x;
  int stride = gridDim.x * blockDim.x;
  for (; i < n4; i += stride) {
    float4 f = ((const float4*)src)[i];
    ushort4 h;
    h.x = f2bf(f.x); h.y = f2bf(f.y); h.z = f2bf(f.z); h.w = f2bf(f.w);
    ((ushort4*)dst)[i] = h;
  }
}

// ---------------------------------------------------------------------------
// Stage 2/4: C = oscale * (X(MxK bf16) @ W(NxK bf16)^T + bias)
// One wave computes a 32x64 tile (two 16-row A tiles share every B operand):
// 8 WMMAs per 12 b128 loads per 32-wide K step.
// MODE 0: out bf16 at [B,H,S,HD]   (Q/K head layout; Q passes oscale=1/8)
// MODE 1: out bf16 at [B,H,HD,S]   (V transposed, for contiguous P@V B-operand)
// MODE 3: out f32  row-major [M,N] (final projection -> d_out)
// ---------------------------------------------------------------------------
template<int MODE>
__global__ __launch_bounds__(128)
void gemm_bias_wmma(const unsigned short* __restrict__ X,
                    const unsigned short* __restrict__ W,
                    const float* __restrict__ bias,
                    void* __restrict__ out,
                    int M, int N, int K, float oscale) {
  const int lane = threadIdx.x & 31;
  const int wave = threadIdx.x >> 5;
  const int half = lane >> 4;       // lane-group within wave32
  const int l16  = lane & 15;
  const int m0 = blockIdx.x * 128 + wave * 32;
  const int n0 = blockIdx.y * 64;

  // A layout (16-bit, 16x32): lane<16 -> K = kk+{0..7, 16..23}; lane>=16 -> +8
  const unsigned short* Xrow0 = X + (size_t)(m0 + l16) * K + 8 * half;
  const unsigned short* Xrow1 = Xrow0 + (size_t)16 * K;
  // B layout (16-bit, 32x16): lane holds column N=l16 (= W row), K-range by half
  const unsigned short* Wb    = W + (size_t)(n0 + l16) * K + 8 * half;

  v8f acc[8];
#pragma unroll
  for (int t = 0; t < 8; ++t) acc[t] = vzero8();

  for (int kk = 0; kk < K; kk += 32) {
    ARegs a0, a1;
    a0.u[0] = *(const uint4*)(Xrow0 + kk);
    a0.u[1] = *(const uint4*)(Xrow0 + kk + 16);
    a1.u[0] = *(const uint4*)(Xrow1 + kk);
    a1.u[1] = *(const uint4*)(Xrow1 + kk + 16);
#pragma unroll
    for (int t = 0; t < 4; ++t) {
      ARegs b;
      const unsigned short* wp = Wb + (size_t)t * 16 * K + kk;
      b.u[0] = *(const uint4*)(wp);             // K = kk + 8*half + 0..7
      b.u[1] = *(const uint4*)(wp + 16);        // K = kk + 16 + 8*half + 0..7
      acc[t]     = wmma_bf16(a0, b, acc[t]);
      acc[4 + t] = wmma_bf16(a1, b, acc[4 + t]);
    }
  }

  // C/D layout: VGPR v, lanes 0-15 -> row +v; lanes 16-31 -> row +v+8
#pragma unroll
  for (int half_m = 0; half_m < 2; ++half_m) {
#pragma unroll
    for (int t = 0; t < 4; ++t) {
      const int col = n0 + t * 16 + l16;
      const float bv = bias[col];
#pragma unroll
      for (int v = 0; v < 8; ++v) {
        const int row = m0 + half_m * 16 + v + 8 * half;
        const float val = (acc[half_m * 4 + t][v] + bv) * oscale;
        if (MODE == 3) {
          ((float*)out)[(size_t)row * N + col] = val;
        } else {
          const int bidx = row >> 11;             // / S
          const int s    = row & (SS - 1);
          const int h    = col >> 6;              // / HD
          const int hd   = col & (HDD - 1);
          const size_t idx = (MODE == 0)
            ? (((size_t)(bidx * HH + h) * SS + s) * HDD + hd)
            : (((size_t)(bidx * HH + h) * HDD + hd) * SS + s);
          ((unsigned short*)out)[idx] = f2bf(val);
        }
      }
    }
  }
}

// ---------------------------------------------------------------------------
// Stage 3: causal flash attention. One wave per 16-row Q tile.
// Qh (pre-scaled by 1/sqrt(HD)), Kh: [B,H,S,HD] bf16. Vt: [B,H,HD,S] bf16.
// O: [B*S, D] bf16.
// Loop split: all-but-one 32-key blocks are mask-free; exactly one diagonal
// block needs causal cndmasks (straight-line tail).
// ---------------------------------------------------------------------------
template<bool MASKED>
__device__ __forceinline__ void attn_step(
    int kb, int q0, int half, int l16,
    const unsigned short* __restrict__ Kp,
    const unsigned short* __restrict__ Vp,
    const ARegs& aq0, const ARegs& aq1,
    v8f& o0, v8f& o1, v8f& o2, v8f& o3,
    float* mrow, float* lrow,
    unsigned short (*pb)[40]) {
  // prefetch next K / V^T block into cache while this step computes
  __builtin_prefetch(Kp + (size_t)(kb + 32 + 16 * half + l16) * HDD, 0, 1);
  __builtin_prefetch(Vp + (size_t)(16 * half + l16) * SS + kb + 32, 0, 1);
  __builtin_prefetch(Vp + (size_t)(32 + 16 * half + l16) * SS + kb + 32, 0, 1);

  // ---- scores: S = Q @ K^T (two 16-key tiles, two feature chunks each) ----
  v8f s0 = vzero8(), s1 = vzero8();
  {
    const unsigned short* Kr0 = Kp + (size_t)(kb + l16) * HDD + 16 * half;
    ARegs b;
    b.u[0] = *(const uint4*)(Kr0);      b.u[1] = *(const uint4*)(Kr0 + 8);
    s0 = wmma_bf16(aq0, b, s0);
    b.u[0] = *(const uint4*)(Kr0 + 32); b.u[1] = *(const uint4*)(Kr0 + 40);
    s0 = wmma_bf16(aq1, b, s0);
    const unsigned short* Kr1 = Kr0 + (size_t)16 * HDD;
    b.u[0] = *(const uint4*)(Kr1);      b.u[1] = *(const uint4*)(Kr1 + 8);
    s1 = wmma_bf16(aq0, b, s1);
    b.u[0] = *(const uint4*)(Kr1 + 32); b.u[1] = *(const uint4*)(Kr1 + 40);
    s1 = wmma_bf16(aq1, b, s1);
  }

  // ---- online softmax (per row = per C-layout VGPR within a 16-lane half) ----
  float alpha[8], p0[8], p1[8];
#pragma unroll
  for (int v = 0; v < 8; ++v) {
    float sv0 = s0[v];
    float sv1 = s1[v];
    if (MASKED) {
      const int row = q0 + v + 8 * half;
      if (kb + l16 > row)      sv0 = -1.0e30f;    // causal mask (diagonal block)
      if (kb + 16 + l16 > row) sv1 = -1.0e30f;
    }
    float rm = fmaxf(sv0, sv1);
    rm = fmaxf(rm, __shfl_xor(rm, 1, 16));
    rm = fmaxf(rm, __shfl_xor(rm, 2, 16));
    rm = fmaxf(rm, __shfl_xor(rm, 4, 16));
    rm = fmaxf(rm, __shfl_xor(rm, 8, 16));
    const float mn = fmaxf(mrow[v], rm);
    const float al = __expf(mrow[v] - mn);
    const float e0 = __expf(sv0 - mn);
    const float e1 = __expf(sv1 - mn);
    float rs = e0 + e1;
    rs += __shfl_xor(rs, 1, 16);
    rs += __shfl_xor(rs, 2, 16);
    rs += __shfl_xor(rs, 4, 16);
    rs += __shfl_xor(rs, 8, 16);
    lrow[v] = lrow[v] * al + rs;
    mrow[v] = mn;
    alpha[v] = al; p0[v] = e0; p1[v] = e1;
  }
#pragma unroll
  for (int v = 0; v < 8; ++v) {
    o0[v] *= alpha[v]; o1[v] *= alpha[v]; o2[v] *= alpha[v]; o3[v] *= alpha[v];
  }

  // ---- P: C-layout -> A-layout via per-wave padded LDS tile ----
#pragma unroll
  for (int v = 0; v < 8; ++v) {
    pb[v + 8 * half][l16]      = f2bf(p0[v]);
    pb[v + 8 * half][16 + l16] = f2bf(p1[v]);
  }
  asm volatile("s_wait_dscnt 0" ::: "memory");
  ARegs ap;
  ap.u[0] = *(const uint4*)&pb[l16][16 * half];       // cols 16h..16h+7
  ap.u[1] = *(const uint4*)&pb[l16][16 * half + 8];   // cols 16h+8..+7

  // ---- O += P @ V  (Vt rows are contiguous B-operand columns) ----
  const unsigned short* Vr = Vp + (size_t)l16 * SS + kb + 16 * half;
  ARegs b;
  b.u[0] = *(const uint4*)(Vr + (size_t)0 * 16 * SS);
  b.u[1] = *(const uint4*)(Vr + (size_t)0 * 16 * SS + 8);
  o0 = wmma_bf16(ap, b, o0);
  b.u[0] = *(const uint4*)(Vr + (size_t)1 * 16 * SS);
  b.u[1] = *(const uint4*)(Vr + (size_t)1 * 16 * SS + 8);
  o1 = wmma_bf16(ap, b, o1);
  b.u[0] = *(const uint4*)(Vr + (size_t)2 * 16 * SS);
  b.u[1] = *(const uint4*)(Vr + (size_t)2 * 16 * SS + 8);
  o2 = wmma_bf16(ap, b, o2);
  b.u[0] = *(const uint4*)(Vr + (size_t)3 * 16 * SS);
  b.u[1] = *(const uint4*)(Vr + (size_t)3 * 16 * SS + 8);
  o3 = wmma_bf16(ap, b, o3);
}

__global__ __launch_bounds__(128)
void attn_fwd(const unsigned short* __restrict__ Qh,
              const unsigned short* __restrict__ Kh,
              const unsigned short* __restrict__ Vt,
              unsigned short* __restrict__ O) {
  __shared__ __align__(16) unsigned short pbuf[4][16][40];  // 80B pitch: bank-safe

  const int lane = threadIdx.x & 31;
  const int wave = threadIdx.x >> 5;
  const int half = lane >> 4;
  const int l16  = lane & 15;
  const int bh   = blockIdx.y;          // b*H + h
  const int bidx = bh >> 4;
  const int h    = bh & 15;
  const int q0   = blockIdx.x * 64 + wave * 16;

  const unsigned short* Qp = Qh + (size_t)bh * SS * HDD;
  const unsigned short* Kp = Kh + (size_t)bh * SS * HDD;
  const unsigned short* Vp = Vt + (size_t)bh * HDD * SS;

  // Q tile 16x64 -> two A-operands (feature chunks 0..31, 32..63), kept in regs.
  ARegs aq0, aq1;
  {
    const unsigned short* Qrow = Qp + (size_t)(q0 + l16) * HDD + 16 * half;
    aq0.u[0] = *(const uint4*)(Qrow +  0);
    aq0.u[1] = *(const uint4*)(Qrow +  8);
    aq1.u[0] = *(const uint4*)(Qrow + 32);
    aq1.u[1] = *(const uint4*)(Qrow + 40);
  }

  v8f o0 = vzero8(), o1 = vzero8(), o2 = vzero8(), o3 = vzero8();
  float mrow[8], lrow[8];
#pragma unroll
  for (int v = 0; v < 8; ++v) { mrow[v] = -1.0e30f; lrow[v] = 0.0f; }

  // Full (mask-free) blocks: kb + 31 <= q0 guaranteed for kb < fullEnd.
  const int fullEnd = (q0 >> 5) << 5;
  for (int kb = 0; kb < fullEnd; kb += 32) {
    attn_step<false>(kb, q0, half, l16, Kp, Vp, aq0, aq1,
                     o0, o1, o2, o3, mrow, lrow, pbuf[wave]);
  }
  // Exactly one diagonal block needs the causal mask.
  attn_step<true>(fullEnd, q0, half, l16, Kp, Vp, aq0, aq1,
                  o0, o1, o2, o3, mrow, lrow, pbuf[wave]);

  // ---- epilogue: normalize and write O as [B*S, D] bf16 ----
#pragma unroll
  for (int v = 0; v < 8; ++v) {
    const float inv = 1.0f / lrow[v];
    const int grow = bidx * SS + q0 + v + 8 * half;
    const size_t base = (size_t)grow * DD + (size_t)h * HDD;
    O[base +  0 + l16] = f2bf(o0[v] * inv);
    O[base + 16 + l16] = f2bf(o1[v] * inv);
    O[base + 32 + l16] = f2bf(o2[v] * inv);
    O[base + 48 + l16] = f2bf(o3[v] * inv);
  }
}

// ---------------------------------------------------------------------------
// Host-side orchestration (graph-capture safe: launches only)
// ---------------------------------------------------------------------------
extern "C" void kernel_launch(void* const* d_in, const int* in_sizes, int n_in,
                              void* d_out, int out_size, void* d_ws, size_t ws_size,
                              hipStream_t stream) {
  const float* q  = (const float*)d_in[0];
  const float* k  = (const float*)d_in[1];
  const float* v  = (const float*)d_in[2];
  // d_in[3] = mask: causal, computed analytically in-kernel
  const float* Wq = (const float*)d_in[4];
  const float* bq = (const float*)d_in[5];
  const float* Wk = (const float*)d_in[6];
  const float* bk = (const float*)d_in[7];
  const float* Wv = (const float*)d_in[8];
  const float* bv = (const float*)d_in[9];
  const float* Wo = (const float*)d_in[10];
  const float* bo = (const float*)d_in[11];

  const size_t MSZ = (size_t)BB * SS * DD;  // 8,388,608 activations
  const size_t WSZ = (size_t)DD * DD;       // 1,048,576 weights

  unsigned short* ws  = (unsigned short*)d_ws;
  unsigned short* xq  = ws;                 // bf16 q   (reused later for O)
  unsigned short* xk  = xq  + MSZ;
  unsigned short* xv  = xk  + MSZ;
  unsigned short* wqb = xv  + MSZ;
  unsigned short* wkb = wqb + WSZ;
  unsigned short* wvb = wkb + WSZ;
  unsigned short* wob = wvb + WSZ;
  unsigned short* qh  = wob + WSZ;          // [B,H,S,HD], pre-scaled by 1/8
  unsigned short* kh  = qh  + MSZ;          // [B,H,S,HD]
  unsigned short* vt  = kh  + MSZ;          // [B,H,HD,S]
  unsigned short* ob  = xq;                 // attention output, reuse xq

  const int M = BB * SS, N = DD, K = DD;

  // Stage 1: precision conversion
  cvt_f32_bf16<<<2048, 256, 0, stream>>>(q,  xq,  (int)(MSZ / 4));
  cvt_f32_bf16<<<2048, 256, 0, stream>>>(k,  xk,  (int)(MSZ / 4));
  cvt_f32_bf16<<<2048, 256, 0, stream>>>(v,  xv,  (int)(MSZ / 4));
  cvt_f32_bf16<<<1024, 256, 0, stream>>>(Wq, wqb, (int)(WSZ / 4));
  cvt_f32_bf16<<<1024, 256, 0, stream>>>(Wk, wkb, (int)(WSZ / 4));
  cvt_f32_bf16<<<1024, 256, 0, stream>>>(Wv, wvb, (int)(WSZ / 4));
  cvt_f32_bf16<<<1024, 256, 0, stream>>>(Wo, wob, (int)(WSZ / 4));

  // Stage 2: Q/K/V projections (WMMA GEMMs) into head layouts.
  // Q is pre-scaled by 1/sqrt(HD) (linearity) so attention has no per-step mul.
  dim3 gGrid(M / 128, N / 64);              // (64, 16)
  gemm_bias_wmma<0><<<gGrid, 128, 0, stream>>>(xq, wqb, bq, qh, M, N, K, 0.125f);
  gemm_bias_wmma<0><<<gGrid, 128, 0, stream>>>(xk, wkb, bk, kh, M, N, K, 1.0f);
  gemm_bias_wmma<1><<<gGrid, 128, 0, stream>>>(xv, wvb, bv, vt, M, N, K, 1.0f);

  // Stage 3: causal flash attention
  attn_fwd<<<dim3(SS / 64, BB * HH), 128, 0, stream>>>(qh, kh, vt, ob);

  // Stage 4: output projection -> f32 d_out
  gemm_bias_wmma<3><<<gGrid, 128, 0, stream>>>(ob, wob, bo, d_out, M, N, K, 1.0f);
}
